// SparseEquivariantEdgeAttention_43499428774422
// MI455X (gfx1250) — compile-verified, matrix operand order checked by reading, not run on hardware
//
#include <hip/hip_runtime.h>

#define E_N   8192
#define N_NODES 2048
#define HID   128
#define NRAD  64
#define KNN   32
#define DIN   321          // 2*HID + NRAD + 1
#define DINP  352          // padded to 11 k-tiles of 32
#define CUTF  10.0f
#define LN_EPS 1e-5f

typedef __attribute__((ext_vector_type(16))) _Float16 v16h;
typedef __attribute__((ext_vector_type(8)))  _Float16 v8h;
typedef __attribute__((ext_vector_type(8)))  float    v8f;

// ---------------------------------------------------------------------------
// WMMA f16 16x16x32 fragment loader (CDNA5 ISA 7.12.2).
// Per lane, a fragment is two contiguous 16-byte runs along K:
//   elems [0..7]  = P[row][k0 + 8*half     .. +7]
//   elems [8..15] = P[row][k0 + 16 + 8*half.. +7]
// Works for A tiles (row = M) and, with P = W^T [N x K], for B tiles (row = N).
// All call sites guarantee 16-byte alignment.
// ---------------------------------------------------------------------------
__device__ __forceinline__ v16h load_frag16(const _Float16* P, int ld,
                                            int row0, int k0, int lane) {
  const int row  = row0 + (lane & 15);
  const int half = lane >> 4;
  const _Float16* p = P + (size_t)row * ld + k0 + half * 8;
  v8h lo = *(const v8h*)(p);
  v8h hi = *(const v8h*)(p + 16);
  v16h out;
#pragma unroll
  for (int e = 0; e < 8; ++e) { out[e] = lo[e]; out[8 + e] = hi[e]; }
  return out;
}

__device__ __forceinline__ float siluf(float x) {
  return x / (1.0f + __expf(-x));
}

// ---------------------------------------------------------------------------
// 1) one-time cast: activations -> f16; weights -> TRANSPOSED f16 (W^T [N x K])
//    so B fragments coalesce into b128 loads. W1^T padded to K=352.
// ---------------------------------------------------------------------------
__global__ __launch_bounds__(256) void k_cast(
    const float* __restrict__ Wq, const float* __restrict__ Wk,
    const float* __restrict__ Wv, const float* __restrict__ W2,
    const float* __restrict__ Wg1, const float* __restrict__ Wo,
    const float* __restrict__ W1, const float* __restrict__ ef,
    _Float16* __restrict__ WqT, _Float16* __restrict__ WkT,
    _Float16* __restrict__ WvT, _Float16* __restrict__ W2T,
    _Float16* __restrict__ Wg1T, _Float16* __restrict__ WoT,
    _Float16* __restrict__ W1T, _Float16* __restrict__ efh) {
  int t = blockIdx.x * 256 + threadIdx.x;
  if (t < HID * HID) {           // t = n*128 + k ; read W[k][n]
    int n = t >> 7, k = t & 127;
    float wq = Wq[k * HID + n], wk = Wk[k * HID + n], wv = Wv[k * HID + n];
    float w2 = W2[k * HID + n], wg = Wg1[k * HID + n], wo = Wo[k * HID + n];
    WqT[t]  = (_Float16)wq;  WkT[t]  = (_Float16)wk;  WvT[t] = (_Float16)wv;
    W2T[t]  = (_Float16)w2;  Wg1T[t] = (_Float16)wg;  WoT[t] = (_Float16)wo;
  }
  if (t < HID * DINP) {          // t = n*352 + k ; read W1[k][n], zero-pad k>=321
    int n = t / DINP, k = t % DINP;
    W1T[t] = (k < DIN) ? (_Float16)W1[k * HID + n] : (_Float16)0.0f;
  }
  if (t < E_N * HID) efh[t] = (_Float16)ef[t];
}

// ---------------------------------------------------------------------------
// 2) edge centers
// ---------------------------------------------------------------------------
__global__ __launch_bounds__(256) void k_centers(
    const float* __restrict__ nc, const int* __restrict__ edge_index,
    float* __restrict__ cen) {
  int i = blockIdx.x * 256 + threadIdx.x;
  if (i >= E_N) return;
  int a = edge_index[i];
  int b = edge_index[E_N + i];
#pragma unroll
  for (int c = 0; c < 3; ++c)
    cen[i * 3 + c] = 0.5f * (nc[a * 3 + c] + nc[b * 3 + c]);
}

// ---------------------------------------------------------------------------
// 3) brute-force KNN: one block per edge row; full d2 row in LDS,
//    extract 32 minima (sorted ascending, matching top_k(-d2))
// ---------------------------------------------------------------------------
__global__ __launch_bounds__(256) void k_knn(const float* __restrict__ cen,
                                             int* __restrict__ idx,
                                             float* __restrict__ dist) {
  __shared__ float d2row[E_N];   // 32 KB
  __shared__ float redv[256];
  __shared__ int   redi[256];
  const int i = blockIdx.x;
  const int t = threadIdx.x;
  const float cx = cen[3 * i], cy = cen[3 * i + 1], cz = cen[3 * i + 2];
  for (int j = t; j < E_N; j += 256) {
    float dx = cx - cen[3 * j];
    float dy = cy - cen[3 * j + 1];
    float dz = cz - cen[3 * j + 2];
    d2row[j] = dx * dx + dy * dy + dz * dz;
  }
  __syncthreads();
  for (int kk = 0; kk < KNN; ++kk) {
    float bv = 3.0e38f; int bi = 0;
    for (int j = t; j < E_N; j += 256) {
      float v = d2row[j];
      if (v < bv) { bv = v; bi = j; }
    }
    redv[t] = bv; redi[t] = bi;
    __syncthreads();
    for (int s = 128; s > 0; s >>= 1) {
      if (t < s && redv[t + s] < redv[t]) {
        redv[t] = redv[t + s]; redi[t] = redi[t + s];
      }
      __syncthreads();
    }
    if (t == 0) {
      int j = redi[0];
      idx[i * KNN + kk]  = j;
      dist[i * KNN + kk] = sqrtf(fmaxf(redv[0], 0.0f));
      d2row[j] = 3.0e38f;
    }
    __syncthreads();
  }
}

// ---------------------------------------------------------------------------
// 4) QKV projection: q/k/v = efh @ W   (WMMA; B from W^T)
// ---------------------------------------------------------------------------
__global__ __launch_bounds__(128) void k_gemm_qkv(
    const _Float16* __restrict__ efh, const _Float16* __restrict__ WqT,
    const _Float16* __restrict__ WkT, const _Float16* __restrict__ WvT,
    float* __restrict__ qf, float* __restrict__ kf, float* __restrict__ vf,
    _Float16* __restrict__ khO, _Float16* __restrict__ vhO) {
  const int lane = threadIdx.x & 31, wave = threadIdx.x >> 5;
  const int mt = blockIdx.x;
  const int nt = blockIdx.y * 4 + wave;
  const int which = blockIdx.z;
  const _Float16* WT = (which == 0) ? WqT : (which == 1 ? WkT : WvT);
  v8f c = {};
#pragma unroll
  for (int kt = 0; kt < 4; ++kt) {
    v16h a = load_frag16(efh, HID, mt * 16, kt * 32, lane);
    v16h b = load_frag16(WT, HID, nt * 16, kt * 32, lane);
    c = __builtin_amdgcn_wmma_f32_16x16x32_f16(false, a, false, b,
                                               (short)0, c, false, false);
  }
  const int n  = nt * 16 + (lane & 15);
  const int mb = mt * 16 + 8 * (lane >> 4);
  float* Of = (which == 0) ? qf : (which == 1 ? kf : vf);
#pragma unroll
  for (int j = 0; j < 8; ++j) {
    float x = c[j];
    Of[(mb + j) * HID + n] = x;
    if (which == 1) khO[(mb + j) * HID + n] = (_Float16)x;
    if (which == 2) vhO[(mb + j) * HID + n] = (_Float16)x;
  }
}

// ---------------------------------------------------------------------------
// 5) gate hidden: g1 = silu(vh @ Wg1 + bg1)  (WMMA -> f16)
// ---------------------------------------------------------------------------
__global__ __launch_bounds__(128) void k_gemm_g1(
    const _Float16* __restrict__ vh, const _Float16* __restrict__ Wg1T,
    const float* __restrict__ bg1, _Float16* __restrict__ g1h) {
  const int lane = threadIdx.x & 31, wave = threadIdx.x >> 5;
  const int mt = blockIdx.x;
  const int nt = blockIdx.y * 4 + wave;
  v8f c = {};
#pragma unroll
  for (int kt = 0; kt < 4; ++kt) {
    v16h a = load_frag16(vh, HID, mt * 16, kt * 32, lane);
    v16h b = load_frag16(Wg1T, HID, nt * 16, kt * 32, lane);
    c = __builtin_amdgcn_wmma_f32_16x16x32_f16(false, a, false, b,
                                               (short)0, c, false, false);
  }
  const int n  = nt * 16 + (lane & 15);
  const int mb = mt * 16 + 8 * (lane >> 4);
  const float bb = bg1[n];
#pragma unroll
  for (int j = 0; j < 8; ++j)
    g1h[(mb + j) * HID + n] = (_Float16)siluf(c[j] + bb);
}

// ---------------------------------------------------------------------------
// 6) per-edge gate scalar
// ---------------------------------------------------------------------------
__global__ __launch_bounds__(256) void k_gates(
    const _Float16* __restrict__ g1h, const float* __restrict__ Wg2,
    const float* __restrict__ bg2, float* __restrict__ gates) {
  int i = blockIdx.x * 256 + threadIdx.x;
  if (i >= E_N) return;
  float s = bg2[0];
  for (int d = 0; d < HID; ++d) s += (float)g1h[i * HID + d] * Wg2[d];
  gates[i] = 1.0f / (1.0f + __expf(-s));
}

// ---------------------------------------------------------------------------
// 7) fused attention MLP + softmax + weighted sums (one block per edge)
// ---------------------------------------------------------------------------
__global__ __launch_bounds__(128) void k_attn(
    const float* __restrict__ ef, const float* __restrict__ ec,
    const float* __restrict__ cen, const int* __restrict__ idx,
    const float* __restrict__ dist, const float* __restrict__ qf,
    const float* __restrict__ kf, const float* __restrict__ vf,
    const _Float16* __restrict__ kh, const _Float16* __restrict__ W1T,
    const float* __restrict__ b1, const _Float16* __restrict__ W2T,
    const float* __restrict__ b2, const float* __restrict__ W3,
    const float* __restrict__ b3, const float* __restrict__ gates,
    const float* __restrict__ centers, const float* __restrict__ widths,
    float* __restrict__ updf, _Float16* __restrict__ updh,
    float* __restrict__ outCoords) {
  __shared__ _Float16 sIn[KNN][DINP];   // 22.5 KB  MLP input tile (704B rows)
  __shared__ _Float16 sH1[KNN][HID];    //  8  KB
  __shared__ float    sH2[KNN][HID];    // 16  KB
  __shared__ float    sQ[HID];
  __shared__ _Float16 sQh[HID];
  __shared__ float    sRed[KNN][4];
  __shared__ int      sJ[KNN];
  __shared__ float    sDist[KNN];
  __shared__ float    sW[KNN];
  __shared__ float    sWG[KNN];
  __shared__ float    sM, sZ;

  const int i = blockIdx.x;
  const int t = threadIdx.x;
  const int lane = t & 31;
  const int wave = t >> 5;

  {
    float qv = qf[i * HID + t];
    sQ[t]  = qv;
    sQh[t] = (_Float16)qv;
  }
  if (t < KNN) { sJ[t] = idx[i * KNN + t]; sDist[t] = dist[i * KNN + t]; }
  __syncthreads();

  // build MLP input: [q | k_j | rbf | dot | zero-pad], vectorized b128 copies
  {
    const int kk = t >> 2, qq = t & 3;             // 4 threads per neighbor row
    const uint4* srcQ = (const uint4*)sQh;                      // 128 f16 = 16 uint4
    const uint4* srcK = (const uint4*)(kh + (size_t)sJ[kk] * HID);
    uint4* dstQ = (uint4*)&sIn[kk][0];
    uint4* dstK = (uint4*)&sIn[kk][HID];
#pragma unroll
    for (int c0 = 0; c0 < 16; c0 += 4) {
      dstQ[c0 + qq] = srcQ[c0 + qq];
      dstK[c0 + qq] = srcK[c0 + qq];
    }
  }
  for (int e = t; e < KNN * NRAD; e += 128) {
    int kk = e >> 6, r = e & 63;
    float d = sDist[kk];
    float val = 0.0f;
    if (d <= CUTF) {
      float x = d - centers[r];
      val = __expf(-widths[r] * x * x);
    }
    sIn[kk][2 * HID + r] = (_Float16)val;
  }
  for (int e = t; e < KNN * (DINP - DIN); e += 128) {
    int kk = e / (DINP - DIN), c = e % (DINP - DIN);
    sIn[kk][DIN + c] = (_Float16)0.0f;
  }
  {  // dot(q, k_j) in f32: 4 partials per row
    int kk = t >> 2, qq = t & 3;
    const float* krow = kf + (size_t)sJ[kk] * HID;
    float p = 0.0f;
    for (int d = qq * 32; d < qq * 32 + 32; ++d) p += sQ[d] * krow[d];
    sRed[kk][qq] = p;
  }
  __syncthreads();
  if (t < KNN)
    sIn[t][2 * HID + NRAD] =
        (_Float16)(sRed[t][0] + sRed[t][1] + sRed[t][2] + sRed[t][3]);
  __syncthreads();

  // GEMM1: [32x352]@[352x128] -> silu -> sH1 (f16)
  for (int step = 0; step < 4; ++step) {
    int tile = wave + 4 * step;            // 16 tiles, 4 waves
    int mt = tile >> 3, nt = tile & 7;
    v8f c = {};
#pragma unroll
    for (int kt = 0; kt < DINP / 32; ++kt) {
      v16h a = load_frag16(&sIn[0][0], DINP, mt * 16, kt * 32, lane);
      v16h b = load_frag16(W1T, DINP, nt * 16, kt * 32, lane);
      c = __builtin_amdgcn_wmma_f32_16x16x32_f16(false, a, false, b,
                                                 (short)0, c, false, false);
    }
    int n  = nt * 16 + (lane & 15);
    int mb = mt * 16 + 8 * (lane >> 4);
    float bb = b1[n];
#pragma unroll
    for (int j = 0; j < 8; ++j)
      sH1[mb + j][n] = (_Float16)siluf(c[j] + bb);
  }
  __syncthreads();

  // GEMM2: [32x128]@[128x128] -> silu -> sH2 (f32)
  for (int step = 0; step < 4; ++step) {
    int tile = wave + 4 * step;
    int mt = tile >> 3, nt = tile & 7;
    v8f c = {};
#pragma unroll
    for (int kt = 0; kt < 4; ++kt) {
      v16h a = load_frag16(&sH1[0][0], HID, mt * 16, kt * 32, lane);
      v16h b = load_frag16(W2T, HID, nt * 16, kt * 32, lane);
      c = __builtin_amdgcn_wmma_f32_16x16x32_f16(false, a, false, b,
                                                 (short)0, c, false, false);
    }
    int n  = nt * 16 + (lane & 15);
    int mb = mt * 16 + 8 * (lane >> 4);
    float bb = b2[n];
#pragma unroll
    for (int j = 0; j < 8; ++j)
      sH2[mb + j][n] = siluf(c[j] + bb);
  }
  __syncthreads();

  // raw score: h2 @ W3 + b3 ; apply cutoff
  {
    int kk = t >> 2, qq = t & 3;
    float p = 0.0f;
    for (int d = qq * 32; d < qq * 32 + 32; ++d) p += sH2[kk][d] * W3[d];
    sRed[kk][qq] = p;
  }
  __syncthreads();
  if (t < KNN) {
    float raw = sRed[t][0] + sRed[t][1] + sRed[t][2] + sRed[t][3] + b3[0];
    sW[t] = (sDist[t] <= CUTF) ? raw : 0.0f;   // scores
  }
  __syncthreads();
  // softmax over full row of E with (E-K) zeros in background
  if (t == 0) {
    float m = 0.0f;
    for (int kk = 0; kk < KNN; ++kk) m = fmaxf(m, sW[kk]);
    sM = m;
  }
  __syncthreads();
  if (t < KNN) sWG[t] = __expf(sW[t] - sM);
  __syncthreads();
  if (t == 0) {
    float Z = (float)(E_N - KNN) * __expf(-sM);
    for (int kk = 0; kk < KNN; ++kk) Z += sWG[kk];
    sZ = Z;
  }
  __syncthreads();
  if (t < KNN) {
    float w = sWG[t] / sZ;
    sW[t]  = w;
    sWG[t] = w * gates[sJ[t]];
  }
  __syncthreads();

  // updated features = ef + sum_k w_k * v_jk
  {
    float acc = ef[i * HID + t];
    for (int kk = 0; kk < KNN; ++kk)
      acc += sW[kk] * vf[(size_t)sJ[kk] * HID + t];
    updf[i * HID + t] = acc;
    updh[i * HID + t] = (_Float16)acc;
  }
  // coordinate update
  if (t < 3) {
    float ci  = cen[i * 3 + t];
    float acc = ec[i * 3 + t];
    for (int kk = 0; kk < KNN; ++kk)
      acc += sWG[kk] * (ci - cen[sJ[kk] * 3 + t]);
    outCoords[i * 3 + t] = acc;
  }
}

// ---------------------------------------------------------------------------
// 8) output projection + residual: x = ef + updated @ Wo + bo
// ---------------------------------------------------------------------------
__global__ __launch_bounds__(128) void k_gemm_out(
    const _Float16* __restrict__ updh, const _Float16* __restrict__ WoT,
    const float* __restrict__ bo, const float* __restrict__ ef,
    float* __restrict__ xf) {
  const int lane = threadIdx.x & 31, wave = threadIdx.x >> 5;
  const int mt = blockIdx.x;
  const int nt = blockIdx.y * 4 + wave;
  v8f c = {};
#pragma unroll
  for (int kt = 0; kt < 4; ++kt) {
    v16h a = load_frag16(updh, HID, mt * 16, kt * 32, lane);
    v16h b = load_frag16(WoT, HID, nt * 16, kt * 32, lane);
    c = __builtin_amdgcn_wmma_f32_16x16x32_f16(false, a, false, b,
                                               (short)0, c, false, false);
  }
  const int n  = nt * 16 + (lane & 15);
  const int mb = mt * 16 + 8 * (lane >> 4);
  const float bb = bo[n];
#pragma unroll
  for (int j = 0; j < 8; ++j)
    xf[(mb + j) * HID + n] = ef[(mb + j) * HID + n] + c[j] + bb;
}

// ---------------------------------------------------------------------------
// 9) LayerNorm (one block per edge)
// ---------------------------------------------------------------------------
__global__ __launch_bounds__(128) void k_ln(const float* __restrict__ xf,
                                            const float* __restrict__ gamma,
                                            const float* __restrict__ beta,
                                            float* __restrict__ out) {
  __shared__ float red[128];
  const int i = blockIdx.x, t = threadIdx.x;
  float x = xf[i * HID + t];
  red[t] = x;
  __syncthreads();
  for (int s = 64; s > 0; s >>= 1) {
    if (t < s) red[t] += red[t + s];
    __syncthreads();
  }
  float mu = red[0] * (1.0f / HID);
  __syncthreads();
  float d = x - mu;
  red[t] = d * d;
  __syncthreads();
  for (int s = 64; s > 0; s >>= 1) {
    if (t < s) red[t] += red[t + s];
    __syncthreads();
  }
  float var = red[0] * (1.0f / HID);
  out[i * HID + t] = d * rsqrtf(var + LN_EPS) * gamma[t] + beta[t];
}

// ---------------------------------------------------------------------------
extern "C" void kernel_launch(void* const* d_in, const int* in_sizes, int n_in,
                              void* d_out, int out_size, void* d_ws,
                              size_t ws_size, hipStream_t stream) {
  (void)in_sizes; (void)n_in; (void)out_size; (void)ws_size;
  const float* ef      = (const float*)d_in[0];
  const float* ec      = (const float*)d_in[1];
  const float* nc      = (const float*)d_in[2];
  const float* Wq      = (const float*)d_in[3];
  const float* Wk      = (const float*)d_in[4];
  const float* Wv      = (const float*)d_in[5];
  const float* W1      = (const float*)d_in[6];
  const float* b1      = (const float*)d_in[7];
  const float* W2      = (const float*)d_in[8];
  const float* b2      = (const float*)d_in[9];
  const float* W3      = (const float*)d_in[10];
  const float* b3      = (const float*)d_in[11];
  const float* Wg1     = (const float*)d_in[12];
  const float* bg1     = (const float*)d_in[13];
  const float* Wg2     = (const float*)d_in[14];
  const float* bg2     = (const float*)d_in[15];
  const float* Wo      = (const float*)d_in[16];
  const float* bo      = (const float*)d_in[17];
  const float* gamma   = (const float*)d_in[18];
  const float* beta    = (const float*)d_in[19];
  const float* centers = (const float*)d_in[20];
  const float* widths  = (const float*)d_in[21];
  const int*   edge_index = (const int*)d_in[22];

  char* ws = (char*)d_ws;
  size_t off = 0;
  auto take = [&](size_t bytes) -> char* {
    char* p = ws + off;
    off = (off + bytes + 255) & ~(size_t)255;
    return p;
  };
  float*    cen   = (float*)take((size_t)E_N * 3 * 4);
  int*      idx   = (int*)take((size_t)E_N * KNN * 4);
  float*    dist  = (float*)take((size_t)E_N * KNN * 4);
  _Float16* efh   = (_Float16*)take((size_t)E_N * HID * 2);
  float*    qf    = (float*)take((size_t)E_N * HID * 4);
  float*    kf    = (float*)take((size_t)E_N * HID * 4);
  float*    vf    = (float*)take((size_t)E_N * HID * 4);
  _Float16* kh    = (_Float16*)take((size_t)E_N * HID * 2);
  _Float16* vh    = (_Float16*)take((size_t)E_N * HID * 2);
  _Float16* g1h   = (_Float16*)take((size_t)E_N * HID * 2);
  float*    gates = (float*)take((size_t)E_N * 4);
  float*    updf  = (float*)take((size_t)E_N * HID * 4);
  _Float16* updh  = (_Float16*)take((size_t)E_N * HID * 2);
  float*    xf    = (float*)take((size_t)E_N * HID * 4);
  _Float16* WqT   = (_Float16*)take((size_t)HID * HID * 2);
  _Float16* WkT   = (_Float16*)take((size_t)HID * HID * 2);
  _Float16* WvT   = (_Float16*)take((size_t)HID * HID * 2);
  _Float16* W2T   = (_Float16*)take((size_t)HID * HID * 2);
  _Float16* Wg1T  = (_Float16*)take((size_t)HID * HID * 2);
  _Float16* WoT   = (_Float16*)take((size_t)HID * HID * 2);
  _Float16* W1T   = (_Float16*)take((size_t)HID * DINP * 2);

  float* outFeat   = (float*)d_out;
  float* outCoords = outFeat + (size_t)E_N * HID;

  k_cast<<<dim3((E_N * HID + 255) / 256), dim3(256), 0, stream>>>(
      Wq, Wk, Wv, W2, Wg1, Wo, W1, ef,
      WqT, WkT, WvT, W2T, Wg1T, WoT, W1T, efh);

  k_centers<<<dim3((E_N + 255) / 256), dim3(256), 0, stream>>>(
      nc, edge_index, cen);

  k_knn<<<dim3(E_N), dim3(256), 0, stream>>>(cen, idx, dist);

  k_gemm_qkv<<<dim3(E_N / 16, 2, 3), dim3(128), 0, stream>>>(
      efh, WqT, WkT, WvT, qf, kf, vf, kh, vh);

  k_gemm_g1<<<dim3(E_N / 16, 2), dim3(128), 0, stream>>>(vh, Wg1T, bg1, g1h);

  k_gates<<<dim3((E_N + 255) / 256), dim3(256), 0, stream>>>(
      g1h, Wg2, bg2, gates);

  k_attn<<<dim3(E_N), dim3(128), 0, stream>>>(
      ef, ec, cen, idx, dist, qf, kf, vf, kh, W1T, b1, W2T, b2, W3, b3,
      gates, centers, widths, updf, updh, outCoords);

  k_gemm_out<<<dim3(E_N / 16, 2), dim3(128), 0, stream>>>(
      updh, WoT, bo, ef, xf);

  k_ln<<<dim3(E_N), dim3(128), 0, stream>>>(xf, gamma, beta, outFeat);
}
